// CustomAttentionBlock_42039139893361
// MI455X (gfx1250) — compile-verified, matrix operand order checked by reading, not run on hardware
//
#include <hip/hip_runtime.h>
#include <hip/hip_bf16.h>
#include <math.h>

// ---------------------------------------------------------------------------
// MI455X (gfx1250) transformer encoder layer with banded rel-pos attention.
// Heavy math through v_wmma_f32_16x16x32_{bf16,f16}; GEMM staging through
// CDNA5 async global->LDS DMA (ASYNCcnt) with double-buffered LDS.
// GEMM: 256x128 block tile, 8 waves x (64x64), 16 WMMA per wave per K-step.
// ---------------------------------------------------------------------------

typedef __attribute__((ext_vector_type(16))) _Float16 v16h;
typedef __attribute__((ext_vector_type(8)))  _Float16 v8h;
typedef __attribute__((ext_vector_type(16))) __bf16   v16bf;
typedef __attribute__((ext_vector_type(8)))  __bf16   v8bf;
typedef __attribute__((ext_vector_type(8)))  float    v8f;

#define B_   4
#define N_   1024
#define D_   768
#define H_   12
#define HD_  64
#define FF_  3072
#define M_   100
#define R_   199   // 2M-1
#define RP_  208   // R_ padded to multiple of 16

static __device__ __forceinline__ v16bf cat_bf(v8bf a, v8bf b) {
  return __builtin_shufflevector(a, b, 0,1,2,3,4,5,6,7,8,9,10,11,12,13,14,15);
}
static __device__ __forceinline__ v16h cat_h(v8h a, v8h b) {
  return __builtin_shufflevector(a, b, 0,1,2,3,4,5,6,7,8,9,10,11,12,13,14,15);
}
static __device__ __forceinline__ v8f wmma_bf(v16bf a, v16bf b, v8f c) {
  return __builtin_amdgcn_wmma_f32_16x16x32_bf16(false, a, false, b, (short)0, c, false, false);
}
static __device__ __forceinline__ v8f wmma_h(v16h a, v16h b, v8f c) {
  return __builtin_amdgcn_wmma_f32_16x16x32_f16(false, a, false, b, (short)0, c, false, false);
}
static __device__ __forceinline__ float rmax16(float v) {
  v = fmaxf(v, __shfl_xor(v, 1, 32));
  v = fmaxf(v, __shfl_xor(v, 2, 32));
  v = fmaxf(v, __shfl_xor(v, 4, 32));
  v = fmaxf(v, __shfl_xor(v, 8, 32));
  return v;
}
static __device__ __forceinline__ float rsum16(float v) {
  v += __shfl_xor(v, 1, 32);
  v += __shfl_xor(v, 2, 32);
  v += __shfl_xor(v, 4, 32);
  v += __shfl_xor(v, 8, 32);
  return v;
}
// CDNA5 async DMA: 16B per lane, global -> LDS, tracked with ASYNCcnt.
static __device__ __forceinline__ void async_b128(const void* g, void* l) {
  unsigned lds = (unsigned)(unsigned long long)l;   // low 32 bits = LDS offset
  asm volatile("global_load_async_to_lds_b128 %0, %1, off"
               :: "v"(lds), "v"(g) : "memory");
}

// --------------------------- fp32 -> bf16 convert ---------------------------
__global__ void cvt_kernel(const float* __restrict__ in, __bf16* __restrict__ out, int n) {
  int i = blockIdx.x * 256 + threadIdx.x;
  if (i < n) out[i] = (__bf16)in[i];
}

// rel_emb (H,R,64) f32 -> (H,RP,64) bf16 scaled by 8 (compensates Qb's 1/8)
__global__ void cvt_emb_kernel(const float* __restrict__ in, __bf16* __restrict__ out) {
  int idx = blockIdx.x * 256 + threadIdx.x;
  if (idx >= H_ * R_ * HD_) return;
  int d = idx & 63;
  int r = (idx >> 6) % R_;
  int h = idx / (R_ * HD_);
  out[((size_t)h * RP_ + r) * HD_ + d] = (__bf16)(in[idx] * 8.0f);
}

// --------------------- generic bf16 WMMA GEMM: C = A * W^T -------------------
// Block tile 256(M) x 128(N), 8 waves of 64x64. K staged 32 at a time through
// async global->LDS DMA into double-buffered LDS; each thread issues exactly
// six async b128s per buffer (4 A + 2 B), so s_wait_asynccnt 6 retires the
// older buffer while the newer one is still in flight.
// epi: 0 = +bias -> f32 ; 1 = +bias+res -> f32 ; 2 = gelu(+bias) -> bf16
__global__ __launch_bounds__(256) void gemm_bf16_kernel(
    const __bf16* __restrict__ A, const __bf16* __restrict__ W,
    const float* __restrict__ bias, const float* __restrict__ res,
    float* __restrict__ outF, __bf16* __restrict__ outB,
    int M, int N, int K, int epi)
{
  __shared__ __bf16 sA[2][256 * 40];
  __shared__ __bf16 sB[2][128 * 40];
  const int tid  = threadIdx.x;
  const int lane = tid & 31;
  const int wid  = tid >> 5;
  const int half = lane >> 4;
  const int lr   = lane & 15;
  const int k0e  = half ? 8 : 0;
  const int m0 = blockIdx.y * 256;
  const int n0 = blockIdx.x * 128;
  const int wm = (wid & 3) * 64;
  const int wn = (wid >> 2) * 64;

  // staging decomposition: 16B chunks, 8 bf16 each; 4 chunks per 32-wide row
  const int crow = tid >> 2;         // base row 0..63
  const int ccol = (tid & 3) * 8;    // col offset in 32-wide K slice

  v8f c[4][4];
#pragma unroll
  for (int i = 0; i < 4; ++i)
#pragma unroll
    for (int j = 0; j < 4; ++j) c[i][j] = (v8f){};

  auto stage = [&](int buf, int k0) {
#pragma unroll
    for (int p = 0; p < 4; ++p) {   // A: 256 rows x 32
      int r = crow + p * 64;
      async_b128(A + (size_t)(m0 + r) * K + k0 + ccol, &sA[buf][r * 40 + ccol]);
    }
#pragma unroll
    for (int p = 0; p < 2; ++p) {   // B: 128 rows x 32 (rows of W)
      int r = crow + p * 64;
      async_b128(W + (size_t)(n0 + r) * K + k0 + ccol, &sB[buf][r * 40 + ccol]);
    }
  };

  const int nk = K / 32;
  stage(0, 0);
  for (int ik = 0; ik < nk; ++ik) {
    const int cur = ik & 1;
    if (ik + 1 < nk) {
      stage(cur ^ 1, (ik + 1) * 32);
      asm volatile("s_wait_asynccnt 0x6" ::: "memory"); // older 6 (cur) retired
    } else {
      asm volatile("s_wait_asynccnt 0x0" ::: "memory");
    }
    __syncthreads();   // everyone's DMA for `cur` has landed in LDS
    v16bf af[4], bf[4];
#pragma unroll
    for (int i = 0; i < 4; ++i) {
      const __bf16* p = &sA[cur][(wm + i * 16 + lr) * 40 + k0e];
      af[i] = cat_bf(*(const v8bf*)p, *(const v8bf*)(p + 16));
    }
#pragma unroll
    for (int j = 0; j < 4; ++j) {
      const __bf16* p = &sB[cur][(wn + j * 16 + lr) * 40 + k0e];
      bf[j] = cat_bf(*(const v8bf*)p, *(const v8bf*)(p + 16));
    }
#pragma unroll
    for (int i = 0; i < 4; ++i)
#pragma unroll
      for (int j = 0; j < 4; ++j)
        c[i][j] = wmma_bf(af[i], bf[j], c[i][j]);
    __syncthreads();   // done reading `cur`; it may be overwritten next iter
  }

  float bn[4];
#pragma unroll
  for (int j = 0; j < 4; ++j) bn[j] = bias[n0 + wn + j * 16 + lr];
#pragma unroll
  for (int i = 0; i < 4; ++i) {
#pragma unroll
    for (int j = 0; j < 4; ++j) {
#pragma unroll
      for (int v = 0; v < 8; ++v) {
        int m = m0 + wm + i * 16 + v + 8 * half;
        int n = n0 + wn + j * 16 + lr;
        float x = c[i][j][v] + bn[j];
        if (epi == 1) x += res[(size_t)m * N + n];
        if (epi == 2) { // exact GELU, bf16 out
          x = 0.5f * x * (1.0f + erff(x * 0.70710678118f));
          outB[(size_t)m * N + n] = (__bf16)x;
        } else {
          outF[(size_t)m * N + n] = x;
        }
      }
    }
  }
}

// -------- repack qkv f32 -> Q(bf16, pre-scaled by 1/8), K(bf16), V^T(f16) ----
__global__ void repack_kernel(const float* __restrict__ qkv, __bf16* __restrict__ Qb,
                              __bf16* __restrict__ Kb, _Float16* __restrict__ Vt)
{
  int idx = blockIdx.x * blockDim.x + threadIdx.x;
  if (idx >= B_ * N_ * D_) return;
  int hd = idx & 63;
  int h  = (idx >> 6) % H_;
  int n  = (idx / D_) % N_;
  int b  = idx / (N_ * D_);
  size_t row = (size_t)b * N_ + n;
  float q = qkv[row * (3 * D_) + h * HD_ + hd];
  float k = qkv[row * (3 * D_) + D_ + h * HD_ + hd];
  float v = qkv[row * (3 * D_) + 2 * D_ + h * HD_ + hd];
  size_t bh = (size_t)(b * H_ + h);
  Qb[(bh * N_ + n) * HD_ + hd] = (__bf16)(q * 0.125f);   // fold 1/sqrt(HD)
  Kb[(bh * N_ + n) * HD_ + hd] = (__bf16)k;
  Vt[(bh * HD_ + hd) * N_ + n] = (_Float16)v;
}

// -------- positional logits P[bh,i,r] via WMMA, faithful flat-reshape --------
// Query row is the scrambled (b,n) = divmod(4*i + b2, N) for head h2 (the
// fairseq reshape). Qb carries 1/8; Eb carries x8, so P is unscaled.
// One wave per (bh, 16-query tile); 13 n-tiles of 16 over RP_=208 (R_=199).
__global__ __launch_bounds__(256) void pos_wmma_kernel(
    const __bf16* __restrict__ Qb, const __bf16* __restrict__ Eb,
    float* __restrict__ P)
{
  const int tid  = threadIdx.x;
  const int lane = tid & 31;
  const int wid  = tid >> 5;
  const int half = lane >> 4;
  const int lr   = lane & 15;
  const int k0e  = half ? 8 : 0;
  const int gw = blockIdx.x * 8 + wid;
  const int bh = gw >> 6;
  const int it = gw & 63;
  const int b2 = bh / H_;
  const int h2 = bh % H_;
  const int i0 = it * 16;

  // A-fragment: scrambled query rows
  const int u  = (i0 + lr) * 4 + b2;
  const int bb = u >> 10, nn = u & 1023;
  const __bf16* qrow = Qb + ((size_t)(bb * H_ + h2) * N_ + nn) * HD_;
  v16bf a0 = cat_bf(*(const v8bf*)(qrow + k0e),      *(const v8bf*)(qrow + k0e + 16));
  v16bf a1 = cat_bf(*(const v8bf*)(qrow + 32 + k0e), *(const v8bf*)(qrow + 32 + k0e + 16));

  for (int nt = 0; nt < RP_ / 16; ++nt) {
    const __bf16* erow = Eb + ((size_t)h2 * RP_ + nt * 16 + lr) * HD_;
    v16bf b0 = cat_bf(*(const v8bf*)(erow + k0e),      *(const v8bf*)(erow + k0e + 16));
    v16bf b1 = cat_bf(*(const v8bf*)(erow + 32 + k0e), *(const v8bf*)(erow + 32 + k0e + 16));
    v8f c = {};
    c = wmma_bf(a0, b0, c);
    c = wmma_bf(a1, b1, c);
    const int r = nt * 16 + lr;
#pragma unroll
    for (int v = 0; v < 8; ++v) {
      int i = i0 + v + 8 * half;
      if (r < R_) P[((size_t)bh * N_ + i) * R_ + r] = c[v];
    }
  }
}

// ------------------- banded flash attention (band = +/-99) -------------------
// One wave per (b,h, 16-query tile). QK^T via bf16 WMMA, online softmax,
// probs->f16 through per-wave LDS, PV via f16 WMMA against V^T.
__global__ __launch_bounds__(256) void attn_kernel(
    const __bf16* __restrict__ Q, const __bf16* __restrict__ Kc,
    const _Float16* __restrict__ Vt, const float* __restrict__ P,
    __bf16* __restrict__ ctx)
{
  __shared__ _Float16 sP[8 * 16 * 40];
  const int tid  = threadIdx.x;
  const int lane = tid & 31;
  const int wid  = tid >> 5;
  const int half = lane >> 4;
  const int lr   = lane & 15;
  const int k0e  = half ? 8 : 0;
  const int gw  = blockIdx.x * 8 + wid;
  const int bh  = gw >> 6;          // N_/16 = 64 query tiles
  const int qt  = gw & 63;
  const int b   = bh / H_;
  const int h   = bh % H_;
  const int qi0 = qt * 16;
  _Float16* myP = &sP[wid * (16 * 40)];

  const __bf16* qrow = Q + ((size_t)bh * N_ + qi0 + lr) * HD_;
  v16bf aq0 = cat_bf(*(const v8bf*)(qrow + k0e),      *(const v8bf*)(qrow + k0e + 16));
  v16bf aq1 = cat_bf(*(const v8bf*)(qrow + 32 + k0e), *(const v8bf*)(qrow + 32 + k0e + 16));

  v8f o0 = {}, o1 = {}, o2 = {}, o3 = {};
  float mrow[8], lrow[8];
#pragma unroll
  for (int v = 0; v < 8; ++v) { mrow[v] = -3.0e38f; lrow[v] = 0.f; }

  int jlo = qi0 - (M_ - 1); if (jlo < 0) jlo = 0; jlo &= ~31;
  int jhi = qi0 + 15 + (M_ - 1); if (jhi > N_ - 1) jhi = N_ - 1;

  for (int j0 = jlo; j0 <= jhi; j0 += 32) {
    const __bf16* kr0 = Kc + ((size_t)bh * N_ + j0 + lr) * HD_;
    const __bf16* kr1 = Kc + ((size_t)bh * N_ + j0 + 16 + lr) * HD_;
    v16bf bk00 = cat_bf(*(const v8bf*)(kr0 + k0e),      *(const v8bf*)(kr0 + k0e + 16));
    v16bf bk01 = cat_bf(*(const v8bf*)(kr0 + 32 + k0e), *(const v8bf*)(kr0 + 32 + k0e + 16));
    v16bf bk10 = cat_bf(*(const v8bf*)(kr1 + k0e),      *(const v8bf*)(kr1 + k0e + 16));
    v16bf bk11 = cat_bf(*(const v8bf*)(kr1 + 32 + k0e), *(const v8bf*)(kr1 + 32 + k0e + 16));
    v8f s0 = {}, s1 = {};
    s0 = wmma_bf(aq0, bk00, s0);
    s0 = wmma_bf(aq1, bk01, s0);
    s1 = wmma_bf(aq0, bk10, s1);
    s1 = wmma_bf(aq1, bk11, s1);

#pragma unroll
    for (int v = 0; v < 8; ++v) {
      int i   = qi0 + v + 8 * half;
      int ja  = j0 + lr;
      int jb  = j0 + 16 + lr;
      int ra  = ja - i + (M_ - 1);
      int rb  = jb - i + (M_ - 1);
      float pa = (ra >= 0 && ra < R_) ? P[((size_t)bh * N_ + i) * R_ + ra] : -1.0e8f;
      float pb = (rb >= 0 && rb < R_) ? P[((size_t)bh * N_ + i) * R_ + rb] : -1.0e8f;
      s0[v] += pa;
      s1[v] += pb;
      float t  = rmax16(fmaxf(s0[v], s1[v]));
      float mn = fmaxf(mrow[v], t);
      float sc = __expf(mrow[v] - mn);
      mrow[v] = mn;
      float e0 = __expf(s0[v] - mn);
      float e1 = __expf(s1[v] - mn);
      lrow[v] = lrow[v] * sc + rsum16(e0 + e1);
      myP[(v + 8 * half) * 40 + lr]      = (_Float16)e0;
      myP[(v + 8 * half) * 40 + 16 + lr] = (_Float16)e1;
      o0[v] *= sc; o1[v] *= sc; o2[v] *= sc; o3[v] *= sc;
    }
    // probs A-fragment (same-wave LDS round trip; DS ops are in-order)
    const _Float16* pp = &myP[lr * 40 + k0e];
    v16h paf = cat_h(*(const v8h*)pp, *(const v8h*)(pp + 16));
    const _Float16* vr0 = Vt + ((size_t)bh * HD_ + 0 * 16 + lr) * N_ + j0;
    const _Float16* vr1 = Vt + ((size_t)bh * HD_ + 1 * 16 + lr) * N_ + j0;
    const _Float16* vr2 = Vt + ((size_t)bh * HD_ + 2 * 16 + lr) * N_ + j0;
    const _Float16* vr3 = Vt + ((size_t)bh * HD_ + 3 * 16 + lr) * N_ + j0;
    v16h vb0 = cat_h(*(const v8h*)(vr0 + k0e), *(const v8h*)(vr0 + k0e + 16));
    v16h vb1 = cat_h(*(const v8h*)(vr1 + k0e), *(const v8h*)(vr1 + k0e + 16));
    v16h vb2 = cat_h(*(const v8h*)(vr2 + k0e), *(const v8h*)(vr2 + k0e + 16));
    v16h vb3 = cat_h(*(const v8h*)(vr3 + k0e), *(const v8h*)(vr3 + k0e + 16));
    o0 = wmma_h(paf, vb0, o0);
    o1 = wmma_h(paf, vb1, o1);
    o2 = wmma_h(paf, vb2, o2);
    o3 = wmma_h(paf, vb3, o3);
  }

#pragma unroll
  for (int v = 0; v < 8; ++v) {
    float inv = 1.0f / lrow[v];
    int i = qi0 + v + 8 * half;
    size_t base = ((size_t)b * N_ + i) * D_ + h * HD_;
    ctx[base + 0 * 16 + lr] = (__bf16)(o0[v] * inv);
    ctx[base + 1 * 16 + lr] = (__bf16)(o1[v] * inv);
    ctx[base + 2 * 16 + lr] = (__bf16)(o2[v] * inv);
    ctx[base + 3 * 16 + lr] = (__bf16)(o3[v] * inv);
  }
}

// --------------------------------- LayerNorm --------------------------------
__global__ __launch_bounds__(256) void ln_kernel(const float* __restrict__ in,
                                                 const float* __restrict__ g,
                                                 const float* __restrict__ be,
                                                 float* __restrict__ outF,
                                                 __bf16* __restrict__ outB)
{
  __shared__ float red[256];
  int row = blockIdx.x, tid = threadIdx.x;
  const float* x = in + (size_t)row * D_;
  float s = 0.f;
  for (int c = tid; c < D_; c += 256) s += x[c];
  red[tid] = s; __syncthreads();
  for (int st = 128; st > 0; st >>= 1) { if (tid < st) red[tid] += red[tid + st]; __syncthreads(); }
  float mu = red[0] / D_;
  __syncthreads();
  float s2 = 0.f;
  for (int c = tid; c < D_; c += 256) { float d = x[c] - mu; s2 += d * d; }
  red[tid] = s2; __syncthreads();
  for (int st = 128; st > 0; st >>= 1) { if (tid < st) red[tid] += red[tid + st]; __syncthreads(); }
  float rstd = rsqrtf(red[0] / D_ + 1e-5f);
  for (int c = tid; c < D_; c += 256) {
    float y = (x[c] - mu) * rstd * g[c] + be[c];
    outF[(size_t)row * D_ + c] = y;
    if (outB) outB[(size_t)row * D_ + c] = (__bf16)y;
  }
}

// --------------------------------- launcher ---------------------------------
extern "C" void kernel_launch(void* const* d_in, const int* in_sizes, int n_in,
                              void* d_out, int out_size, void* d_ws, size_t ws_size,
                              hipStream_t stream)
{
  const float* src    = (const float*)d_in[0];
  const float* qkv_w  = (const float*)d_in[1];
  const float* qkv_b  = (const float*)d_in[2];
  const float* rel    = (const float*)d_in[3];
  const float* proj_w = (const float*)d_in[4];
  const float* proj_b = (const float*)d_in[5];
  const float* ln1_g  = (const float*)d_in[6];
  const float* ln1_b  = (const float*)d_in[7];
  const float* fc1_w  = (const float*)d_in[8];
  const float* fc1_b  = (const float*)d_in[9];
  const float* fc2_w  = (const float*)d_in[10];
  const float* fc2_b  = (const float*)d_in[11];
  const float* ln2_g  = (const float*)d_in[12];
  const float* ln2_b  = (const float*)d_in[13];
  (void)in_sizes; (void)n_in; (void)out_size; (void)ws_size;

  char* ws = (char*)d_ws;
  size_t off = 0;
  auto take = [&](size_t bytes) -> char* {
    char* p = ws + off;
    off += (bytes + 255) & ~(size_t)255;
    return p;
  };
  __bf16*   srcB   = (__bf16*)  take((size_t)B_ * N_ * D_ * 2);
  __bf16*   qkvwB  = (__bf16*)  take((size_t)3 * D_ * D_ * 2);
  __bf16*   projwB = (__bf16*)  take((size_t)D_ * D_ * 2);
  __bf16*   fc1wB  = (__bf16*)  take((size_t)FF_ * D_ * 2);
  __bf16*   fc2wB  = (__bf16*)  take((size_t)D_ * FF_ * 2);
  __bf16*   embB   = (__bf16*)  take((size_t)H_ * RP_ * HD_ * 2);
  float*    qkvF   = (float*)   take((size_t)B_ * N_ * 3 * D_ * 4);
  __bf16*   Qb     = (__bf16*)  take((size_t)B_ * N_ * D_ * 2);
  __bf16*   Kb     = (__bf16*)  take((size_t)B_ * N_ * D_ * 2);
  _Float16* Vt     = (_Float16*)take((size_t)B_ * N_ * D_ * 2);
  float*    Pp     = (float*)   take((size_t)B_ * H_ * N_ * R_ * 4);
  __bf16*   ctxB   = (__bf16*)  take((size_t)B_ * N_ * D_ * 2);
  float*    y1     = (float*)   take((size_t)B_ * N_ * D_ * 4);
  float*    xF     = (float*)   take((size_t)B_ * N_ * D_ * 4);
  __bf16*   xB     = (__bf16*)  take((size_t)B_ * N_ * D_ * 2);
  __bf16*   h1B    = (__bf16*)  take((size_t)B_ * N_ * FF_ * 2);
  float*    y2     = (float*)   take((size_t)B_ * N_ * D_ * 4);

  auto cvt = [&](const float* in, __bf16* out, int n) {
    cvt_kernel<<<(n + 255) / 256, 256, 0, stream>>>(in, out, n);
  };
  cvt(src,    srcB,  B_ * N_ * D_);
  cvt(qkv_w,  qkvwB, 3 * D_ * D_);
  cvt(proj_w, projwB, D_ * D_);
  cvt(fc1_w,  fc1wB, FF_ * D_);
  cvt(fc2_w,  fc2wB, D_ * FF_);
  cvt_emb_kernel<<<(H_ * R_ * HD_ + 255) / 256, 256, 0, stream>>>(rel, embB);

  // QKV projection
  gemm_bf16_kernel<<<dim3((3 * D_) / 128, (B_ * N_) / 256), 256, 0, stream>>>(
      srcB, qkvwB, qkv_b, nullptr, qkvF, nullptr, B_ * N_, 3 * D_, D_, 0);
  repack_kernel<<<(B_ * N_ * D_ + 255) / 256, 256, 0, stream>>>(qkvF, Qb, Kb, Vt);
  pos_wmma_kernel<<<(B_ * H_ * (N_ / 16)) / 8, 256, 0, stream>>>(Qb, embB, Pp);
  attn_kernel<<<(B_ * H_ * (N_ / 16)) / 8, 256, 0, stream>>>(Qb, Kb, Vt, Pp, ctxB);
  // output projection + residual
  gemm_bf16_kernel<<<dim3(D_ / 128, (B_ * N_) / 256), 256, 0, stream>>>(
      ctxB, projwB, proj_b, src, y1, nullptr, B_ * N_, D_, D_, 1);
  ln_kernel<<<B_ * N_, 256, 0, stream>>>(y1, ln1_g, ln1_b, xF, xB);
  // FFN
  gemm_bf16_kernel<<<dim3(FF_ / 128, (B_ * N_) / 256), 256, 0, stream>>>(
      xB, fc1wB, fc1_b, nullptr, nullptr, h1B, B_ * N_, FF_, D_, 2);
  gemm_bf16_kernel<<<dim3(D_ / 128, (B_ * N_) / 256), 256, 0, stream>>>(
      h1B, fc2wB, fc2_b, xF, y2, nullptr, B_ * N_, D_, FF_, 1);
  ln_kernel<<<B_ * N_, 256, 0, stream>>>(y2, ln2_g, ln2_b, (float*)d_out, nullptr);
}
